// Clusterator_72481868087581
// MI455X (gfx1250) — matrix operand order repaired and checked by reading, not compiled, not run on hardware
//
#include <hip/hip_runtime.h>

typedef __attribute__((ext_vector_type(16))) _Float16 v16h;
typedef __attribute__((ext_vector_type(8)))  _Float16 v8h;
typedef __attribute__((ext_vector_type(4)))  _Float16 v4h;
typedef __attribute__((ext_vector_type(2)))  _Float16 v2h;
typedef __attribute__((ext_vector_type(8)))  float    v8f;

#define NROWS  200000
#define DDIM   128
#define KCL    64
#define TILE   32
#define NTILES (NROWS / TILE)   // 6250, exact
#define NEPS   1e-6f
#define RT_STRIDE 40            // padded LDS row stride (halves), 80B = 16B-aligned
#define DIST_STRIDE 68          // padded dist row stride (floats), 272B = 16B-aligned

__device__ inline v8f zero8() {
  v8f z;
#pragma unroll
  for (int i = 0; i < 8; ++i) z[i] = 0.0f;
  return z;
}

// ---------------- normalize embeds rows -> f16 ----------------
__global__ __launch_bounds__(256) void k_norm_data(const float* __restrict__ embeds,
                                                   _Float16* __restrict__ out) {
  const int lane = threadIdx.x & 31;
  const int wave = threadIdx.x >> 5;
  const int wpb  = blockDim.x >> 5;
  for (int row = blockIdx.x * wpb + wave; row < NROWS; row += gridDim.x * wpb) {
    const float4 v = ((const float4*)(embeds + (size_t)row * DDIM))[lane];
    float s = v.x * v.x + v.y * v.y + v.z * v.z + v.w * v.w;
#pragma unroll
    for (int m = 16; m >= 1; m >>= 1) s += __shfl_xor(s, m, 32);
    const float inv = 1.0f / (sqrtf(s) + NEPS);
    v4h o;
    o[0] = (_Float16)(v.x * inv); o[1] = (_Float16)(v.y * inv);
    o[2] = (_Float16)(v.z * inv); o[3] = (_Float16)(v.w * inv);
    ((v4h*)(out + (size_t)row * DDIM))[lane] = o;
  }
}

// ---------------- zero per-iteration stats ----------------
__global__ void k_zero(float* __restrict__ cm, float* __restrict__ cr) {
  const int t = blockIdx.x * blockDim.x + threadIdx.x;
  if (t < KCL * DDIM) cm[t] = 0.0f;
  if (t < KCL)        cr[t] = 0.0f;
}

// ---------------- normalize mu rows -> f16 ----------------
__global__ __launch_bounds__(128) void k_norm_mu(const float* __restrict__ mu,
                                                 _Float16* __restrict__ muN) {
  __shared__ float red[128];
  const int row = blockIdx.x, t = threadIdx.x;
  const float v = mu[row * DDIM + t];
  red[t] = v * v;
  __syncthreads();
  for (int s = 64; s > 0; s >>= 1) {
    if (t < s) red[t] += red[t + s];
    __syncthreads();
  }
  const float inv = 1.0f / (sqrtf(red[0]) + NEPS);
  muN[row * DDIM + t] = (_Float16)(v * inv);
}

// ---------------- mu = cluster_mean / cluster_r ----------------
__global__ void k_update(const float* __restrict__ cm, const float* __restrict__ cr,
                         float* __restrict__ muOut) {
  const int t = blockIdx.x * blockDim.x + threadIdx.x;
  if (t < KCL * DDIM) muOut[t] = cm[t] / cr[t >> 7];
}

// ---------------- fused iteration kernel ----------------
__global__ __launch_bounds__(256) void k_main(const _Float16* __restrict__ dataF16,
                                              const _Float16* __restrict__ muN,
                                              const int* __restrict__ tempPtr,
                                              float* __restrict__ clusterMean,
                                              float* __restrict__ clusterR,
                                              float* __restrict__ rOut,
                                              int writeR) {
  __shared__ _Float16 sMu[KCL][DDIM];           // 16 KB
  __shared__ _Float16 sData[TILE][DDIM];        // 8 KB  (row-major tile)
  __shared__ _Float16 sDataT[DDIM][RT_STRIDE];  // 10 KB (transposed tile)
  __shared__ float    sDist[TILE][DIST_STRIDE]; // 8.5 KB (padded vs bank conflicts)
  __shared__ _Float16 sRT[KCL][RT_STRIDE];      // 5 KB  (r transposed)
  __shared__ float    sCR[KCL];                 // 256 B

  const int tid  = threadIdx.x;
  const int lane = tid & 31;
  const int w    = tid >> 5;
  const int h    = lane >> 4;   // half-wave id
  const int lm   = lane & 15;   // lane within half

  const float temp = (float)(*tempPtr);

  // stage normalized mu (f16) into LDS
  for (int i = tid; i < KCL * DDIM / 8; i += 256)
    ((v8h*)&sMu[0][0])[i] = ((const v8h*)muN)[i];
  if (tid < KCL) sCR[tid] = 0.0f;

  // matmul1 subtile per wave: rows half mt, cluster tile nt
  const int mt = w & 1;
  const int nt = w >> 1;
  // matmul2 subtiles per wave: cluster tile mt2, dim tiles ntBase..+3
  const int mt2    = w >> 1;
  const int ntBase = (w & 1) * 4;

  // tile-load role: one row-pair segment per thread
  const int rp  = tid >> 4;   // row pair 0..15 -> rows 2rp, 2rp+1
  const int seg = tid & 15;   // 8-half segment within row

  v8f acc2[4];
#pragma unroll
  for (int q = 0; q < 4; ++q) acc2[q] = zero8();

  for (int tile = blockIdx.x; tile < NTILES; tile += gridDim.x) {
    const int row0 = tile * TILE;
    __syncthreads();  // protect LDS reuse vs previous tile's matmul2 reads

    // cooperative load of data tile: row-major copy + packed transpose
    {
      const v8h d0 = ((const v8h*)(dataF16 + (size_t)(row0 + 2 * rp) * DDIM))[seg];
      const v8h d1 = ((const v8h*)(dataF16 + (size_t)(row0 + 2 * rp + 1) * DDIM))[seg];
      *(v8h*)&sData[2 * rp][seg * 8]     = d0;
      *(v8h*)&sData[2 * rp + 1][seg * 8] = d1;
#pragma unroll
      for (int j = 0; j < 8; ++j) {
        v2h p; p[0] = d0[j]; p[1] = d1[j];
        *(v2h*)&sDataT[seg * 8 + j][2 * rp] = p;
      }
      const int nextTile = tile + gridDim.x;
      if (nextTile < NTILES)
        __builtin_prefetch(dataF16 + (size_t)(nextTile * TILE + 2 * rp) * DDIM + seg * 8, 0, 3);
    }
    __syncthreads();

    // ---- matmul1: dist[16x16] = data_tile @ muN^T, K=128 in 4 chunks ----
    v8f acc = zero8();
#pragma unroll
    for (int c = 0; c < 4; ++c) {
      const int k0 = c * 32;
      v16h a, b;
      const v8h a0 = *(const v8h*)&sData[mt * 16 + lm][k0 + h * 8];
      const v8h a1 = *(const v8h*)&sData[mt * 16 + lm][k0 + 16 + h * 8];
      const v8h b0 = *(const v8h*)&sMu[nt * 16 + lm][k0 + h * 8];
      const v8h b1 = *(const v8h*)&sMu[nt * 16 + lm][k0 + 16 + h * 8];
#pragma unroll
      for (int i = 0; i < 8; ++i) { a[i] = a0[i]; a[8 + i] = a1[i]; b[i] = b0[i]; b[8 + i] = b1[i]; }
      acc = __builtin_amdgcn_wmma_f32_16x16x32_f16(false, a, false, b, (short)0, acc, false, false);
    }
#pragma unroll
    for (int v = 0; v < 8; ++v)
      sDist[mt * 16 + h * 8 + v][nt * 16 + lm] = acc[v];
    __syncthreads();

    // ---- softmax over 64 clusters: 8 threads per row, shfl reductions ----
    {
      const int r = tid >> 3;   // 0..31
      const int g = tid & 7;    // column group
      float e[8];
      float mx = -1e30f;
#pragma unroll
      for (int j = 0; j < 8; ++j) {
        e[j] = sDist[r][g * 8 + j];
        mx = fmaxf(mx, e[j]);
      }
      mx = fmaxf(mx, __shfl_xor(mx, 1, 32));
      mx = fmaxf(mx, __shfl_xor(mx, 2, 32));
      mx = fmaxf(mx, __shfl_xor(mx, 4, 32));
      float sum = 0.0f;
#pragma unroll
      for (int j = 0; j < 8; ++j) { e[j] = __expf(temp * (e[j] - mx)); sum += e[j]; }
      sum += __shfl_xor(sum, 1, 32);
      sum += __shfl_xor(sum, 2, 32);
      sum += __shfl_xor(sum, 4, 32);
      const float invs = 1.0f / sum;
#pragma unroll
      for (int j = 0; j < 8; ++j) {
        e[j] *= invs;
        sRT[g * 8 + j][r] = (_Float16)e[j];
      }
      if (writeR) {
        float4* dst = (float4*)(rOut + (size_t)(row0 + r) * KCL + g * 8);
        dst[0] = make_float4(e[0], e[1], e[2], e[3]);
        dst[1] = make_float4(e[4], e[5], e[6], e[7]);
      }
    }
    __syncthreads();

    // cluster_r partial (reads only, disjoint writes)
    if (tid < KCL) {
      float s = 0.0f;
#pragma unroll 8
      for (int r = 0; r < TILE; ++r) s += (float)sRT[tid][r];
      sCR[tid] += s;
    }

    // ---- matmul2: cluster_mean += r^T(64x32) @ data(32x128), K=32 ----
    {
      v16h a2;
      const v8h a0 = *(const v8h*)&sRT[mt2 * 16 + lm][h * 8];
      const v8h a1 = *(const v8h*)&sRT[mt2 * 16 + lm][16 + h * 8];
#pragma unroll
      for (int i = 0; i < 8; ++i) { a2[i] = a0[i]; a2[8 + i] = a1[i]; }
#pragma unroll
      for (int q = 0; q < 4; ++q) {
        const int nt2 = ntBase + q;
        v16h b2;
        const v8h b0 = *(const v8h*)&sDataT[nt2 * 16 + lm][h * 8];
        const v8h b1 = *(const v8h*)&sDataT[nt2 * 16 + lm][16 + h * 8];
#pragma unroll
        for (int i = 0; i < 8; ++i) { b2[i] = b0[i]; b2[8 + i] = b1[i]; }
        acc2[q] = __builtin_amdgcn_wmma_f32_16x16x32_f16(false, a2, false, b2, (short)0, acc2[q], false, false);
      }
    }
  }

  __syncthreads();
  // flush per-workgroup partials
#pragma unroll
  for (int q = 0; q < 4; ++q) {
    const int n = (ntBase + q) * 16 + lm;
#pragma unroll
    for (int v = 0; v < 8; ++v) {
      const int m = mt2 * 16 + h * 8 + v;
      atomicAdd(&clusterMean[m * DDIM + n], acc2[q][v]);
    }
  }
  if (tid < KCL) atomicAdd(&clusterR[tid], sCR[tid]);
}

extern "C" void kernel_launch(void* const* d_in, const int* in_sizes, int n_in,
                              void* d_out, int out_size, void* d_ws, size_t ws_size,
                              hipStream_t stream) {
  (void)in_sizes; (void)n_in; (void)out_size; (void)ws_size;
  const float* embeds = (const float*)d_in[0];
  const float* init   = (const float*)d_in[1];
  const int*   temp   = (const int*)d_in[2];

  char* ws = (char*)d_ws;
  size_t off = 0;
  _Float16* dataF16 = (_Float16*)(ws + off); off += (size_t)NROWS * DDIM * sizeof(_Float16);
  off = (off + 255) & ~(size_t)255;
  float* muBuf = (float*)(ws + off);        off += (size_t)KCL * DDIM * sizeof(float);
  off = (off + 255) & ~(size_t)255;
  _Float16* muN = (_Float16*)(ws + off);    off += (size_t)KCL * DDIM * sizeof(_Float16);
  off = (off + 255) & ~(size_t)255;
  float* cm = (float*)(ws + off);           off += (size_t)KCL * DDIM * sizeof(float);
  off = (off + 255) & ~(size_t)255;
  float* cr = (float*)(ws + off);           off += (size_t)KCL * sizeof(float);

  float* muOutFinal = (float*)d_out;               // [64,128]
  float* rOut       = (float*)d_out + KCL * DDIM;  // [200000,64]

  k_norm_data<<<2048, 256, 0, stream>>>(embeds, dataF16);

  for (int it = 0; it < 11; ++it) {
    const int last = (it == 10);
    const float* muIn = (it == 0) ? init : muBuf;
    k_zero<<<32, 256, 0, stream>>>(cm, cr);
    k_norm_mu<<<64, 128, 0, stream>>>(muIn, muN);
    k_main<<<1250, 256, 0, stream>>>(dataF16, muN, temp, cm, cr, rOut, last);
    k_update<<<32, 256, 0, stream>>>(cm, cr, last ? muOutFinal : muBuf);
  }
}
